// DBSFIAttention_60266981098130
// MI455X (gfx1250) — compile-verified, matrix operand order checked by reading, not run on hardware
//
#include <hip/hip_runtime.h>
#include <hip/hip_bf16.h>
#include <math.h>

typedef _Float16 h16;
typedef __attribute__((ext_vector_type(16))) _Float16 v16h;
typedef __attribute__((ext_vector_type(8)))  _Float16 v8h;
typedef __attribute__((ext_vector_type(2)))  _Float16 h2v;
typedef __attribute__((ext_vector_type(8)))  float    v8f;

#define DEVINL __device__ __forceinline__

constexpr int Bc = 4;         // batch
constexpr int Cc = 256;       // channels
constexpr int Hc = 128, Wd = 128;
constexpr int NP = Hc * Wd;   // 16384 pixels
constexpr int NWIN = 256;     // 16x16 windows of 8x8

DEVINL float sigm(float x){ return 1.f/(1.f+__expf(-x)); }

DEVINL v8f wmma16(v16h a, v16h b, v8f c){
  return __builtin_amdgcn_wmma_f32_16x16x32_f16(false, a, false, b, (short)0, c, false, false);
}

// ---- fragment helpers -------------------------------------------------------
// B-operand fragment-major layout for a 32(K)x16(N) tile:
//   lane = n + 16*hi, hi = (kk>>3)&1 ; slot j = (kk&7) | ((kk>>4)<<3)
DEVINL int bswz(int n, int kk){
  return ((((kk>>3)&1)<<4) + n)*16 + ((kk&7) | ((kk>>4)<<3));
}
DEVINL v16h load_frag16(const h16* p, int lane){   // p 32B-aligned, frag-major
  return *(const v16h*)(p + (lane<<4));
}
// A-operand fragment from row-major [m][ld] tile (ld multiple of 8, 16B-aligned rows)
DEVINL v16h load_arow(const h16* A, int lane, int ld){
  int m = lane & 15;
  int kb = (lane >> 4) << 3;
  const h16* p = A + m*ld;
  v8h lo = *(const v8h*)(p + kb);
  v8h hi = *(const v8h*)(p + 16 + kb);
  v16h r;
#pragma unroll
  for(int j=0;j<8;j++){ r[j]=lo[j]; r[8+j]=hi[j]; }
  return r;
}

// ---------------------------------------------------------------- reductions
__global__ __launch_bounds__(256) void k_pool(const float* __restrict__ x, float* __restrict__ pool){
  int bc = blockIdx.x;
  const float* p = x + (size_t)bc*NP;
  float s=0;
  for(int i=threadIdx.x;i<NP;i+=256) s+=p[i];
  __shared__ float red[256];
  red[threadIdx.x]=s; __syncthreads();
  for(int st=128;st>0;st>>=1){ if(threadIdx.x<st) red[threadIdx.x]+=red[threadIdx.x+st]; __syncthreads(); }
  if(threadIdx.x==0) pool[bc]=red[0]*(1.f/NP);
}

__global__ __launch_bounds__(256) void k_zmax(const float* __restrict__ x, float* __restrict__ z){
  int bc = blockIdx.x;
  const float* p = x + (size_t)bc*NP;
  float s=-3.4e38f;
  for(int i=threadIdx.x;i<NP;i+=256) s=fmaxf(s,p[i]);
  __shared__ float red[256];
  red[threadIdx.x]=s; __syncthreads();
  for(int st=128;st>0;st>>=1){ if(threadIdx.x<st) red[threadIdx.x]=fmaxf(red[threadIdx.x],red[threadIdx.x+st]); __syncthreads(); }
  if(threadIdx.x==0) z[bc]=red[0];
}

__global__ __launch_bounds__(256) void k_xm(const float* __restrict__ x, float* __restrict__ img){
  int idx = blockIdx.x*256+threadIdx.x;      // b*NP + p
  int b = idx / NP, p = idx % NP;
  const float* xp = x + (size_t)b*Cc*NP + p;
  float s=0;
  for(int c=0;c<Cc;c++) s+=xp[(size_t)c*NP];
  img[idx]=s*(1.f/Cc);
}

// ---------------------------------------------------------------- DCT (mirrored-FFT real part)
__global__ __launch_bounds__(256) void k_trig(float* __restrict__ Rt, float* __restrict__ St){
  int idx=blockIdx.x*256+threadIdx.x;        // 128*128
  int k=idx>>7, n=idx&127;
  const float a=6.283185307179586f/256.f;
  int p1=(k*n)&255, p2=(k*(255-n))&255;
  Rt[idx]=cosf(a*(float)p1)+cosf(a*(float)p2);
  St[idx]=sinf(a*(float)p1)+sinf(a*(float)p2);
}

__global__ __launch_bounds__(128) void k_dct1(const float* __restrict__ img, const float* __restrict__ Rt,
                                              const float* __restrict__ St, float* __restrict__ T1, float* __restrict__ T2){
  int bm=blockIdx.x; int b=bm>>7, m=bm&127;
  __shared__ float row[128];
  row[threadIdx.x]=img[(b<<14)+(m<<7)+threadIdx.x];
  __syncthreads();
  int v=threadIdx.x;
  float s1=0,s2=0;
  for(int n=0;n<128;n++){ float iv=row[n]; s1+=iv*Rt[(v<<7)+n]; s2+=iv*St[(v<<7)+n]; }
  T1[(bm<<7)+v]=s1; T2[(bm<<7)+v]=s2;
}

__global__ __launch_bounds__(128) void k_dct2(const float* __restrict__ Rt, const float* __restrict__ St,
                                              const float* __restrict__ T1, const float* __restrict__ T2,
                                              float* __restrict__ dv){
  int bu=blockIdx.x; int b=bu>>7, u=bu&127;
  __shared__ float rr[128], ss[128];
  rr[threadIdx.x]=Rt[(u<<7)+threadIdx.x];
  ss[threadIdx.x]=St[(u<<7)+threadIdx.x];
  __syncthreads();
  int v=threadIdx.x;
  float s=0;
  for(int m=0;m<128;m++){
    int t=(((b<<7)+m)<<7);
    s+=rr[m]*T1[t+v]-ss[m]*T2[t+v];
  }
  dv[(bu<<7)+v]=s;
}

__global__ __launch_bounds__(256) void k_embed(const float* __restrict__ dv, const float* __restrict__ pool,
                                               float* __restrict__ ctx){
  int b=blockIdx.x; const float* d=dv+((size_t)b<<14);
  float s=0;
  for(int i=threadIdx.x;i<NP;i+=256){ float t=fminf(fmaxf(d[i],-10.f),10.f); s+=t*t; }
  __shared__ float red[256];
  red[threadIdx.x]=s; __syncthreads();
  for(int st=128;st>0;st>>=1){ if(threadIdx.x<st) red[threadIdx.x]+=red[threadIdx.x+st]; __syncthreads(); }
  float nn=sqrtf(red[0]);
  float n1=nn+1e-5f;
  float scale=1.f/(n1*(nn/n1+1e-6f));
  int t=threadIdx.x;
  float e=0;
  for(int j=0;j<64;j++){ float q=d[t*64+j]; e+=fminf(fmaxf(q,-10.f),10.f); }
  ctx[b*Cc+t]=pool[b*Cc+t]*(scale*e*(1.f/64.f));
}

__global__ __launch_bounds__(256) void k_mlp_att(const float* __restrict__ ctx, const float* __restrict__ w1,
    const float* __restrict__ b1, const float* __restrict__ w2, const float* __restrict__ b2,
    const float* __restrict__ freq, float* __restrict__ att, float* __restrict__ alpha){
  int b=blockIdx.x;
  __shared__ float cx[256], hid[64];
  cx[threadIdx.x]=ctx[b*Cc+threadIdx.x]; __syncthreads();
  if(threadIdx.x<64){
    float s=b1[threadIdx.x];
    for(int c=0;c<Cc;c++) s+=cx[c]*w1[threadIdx.x*Cc+c];
    hid[threadIdx.x]=fmaxf(s,0.f);
  }
  __syncthreads();
  float s=b2[threadIdx.x];
  for(int h=0;h<64;h++) s+=hid[h]*w2[threadIdx.x*64+h];
  float a=sigm(s);
  att[b*Cc+threadIdx.x]=a;
  alpha[b*Cc+threadIdx.x]=sigm(freq[0])*a;
}

__global__ __launch_bounds__(256) void k_mlp_scarf(const float* __restrict__ z, const float* __restrict__ w1,
    const float* __restrict__ b1, const float* __restrict__ w2, const float* __restrict__ b2, float* __restrict__ sc){
  int b=blockIdx.x;
  __shared__ float zz[256];
  __shared__ float hid[1024];
  zz[threadIdx.x]=z[b*Cc+threadIdx.x]; __syncthreads();
  for(int h=threadIdx.x;h<1024;h+=256){
    float s=b1[h];
    for(int c=0;c<Cc;c++) s+=zz[c]*w1[h*Cc+c];
    hid[h]=fmaxf(s,0.f);
  }
  __syncthreads();
  float s=b2[threadIdx.x];
  for(int h=0;h<1024;h++) s+=hid[h]*w2[threadIdx.x*1024+h];
  sc[b*Cc+threadIdx.x]=sigm(s);
}

// ---------------------------------------------------------------- weight packing (frag-major)
DEVINL void frag_decode(int idx, int ntiles, int& kchunk, int& ntile, int& kk, int& n){
  int j = idx & 15;
  int lane = (idx >> 4) & 31;
  int t2 = idx >> 9;
  ntile = t2 % ntiles;
  kchunk = t2 / ntiles;
  int hi = lane >> 4;
  n = lane & 15;
  kk = (j & 7) | (hi << 3) | ((j >> 3) << 4);
}

__global__ __launch_bounds__(256) void k_pack_cnn(const float* __restrict__ w1, const float* __restrict__ g1,
    const float* __restrict__ bb1, const float* __restrict__ w2, const float* __restrict__ g2,
    const float* __restrict__ bb2, const float* __restrict__ freq, h16* __restrict__ Wcp, float* __restrict__ biasc){
  float wc=sigm(freq[0]); float om=1.f-wc;
  int idx=blockIdx.x*256+threadIdx.x;          // 9*256*256 halves
  int j=idx&15, lane=(idx>>4)&31, t2=idx>>9;
  int ntile=t2&15, t3=t2>>4, kchunk=t3&7, tap=t3>>3;
  int hi=lane>>4, n=lane&15;
  int kk=(j&7)|(hi<<3)|((j>>3)<<4);
  int ci=kchunk*32+kk, co=ntile*16+n;
  float bns=rsqrtf(1.f+1e-5f);
  float v=g1[co]*bns*w1[(co*Cc+ci)*9+tap] + om*g2[co]*bns*w2[(co*Cc+ci)*9+tap];
  Wcp[idx]=(h16)v;
  if(idx<Cc) biasc[idx]=bb1[idx]+om*bb2[idx];
}

__global__ __launch_bounds__(256) void k_pack_s3(const float* __restrict__ w, h16* __restrict__ o){
  int idx=blockIdx.x*256+threadIdx.x;
  int j=idx&15, lane=(idx>>4)&31, t2=idx>>9;
  int ntile=t2&15, t3=t2>>4, kchunk=t3&7, tap=t3>>3;
  int hi=lane>>4, n=lane&15;
  int kk=(j&7)|(hi<<3)|((j>>3)<<4);
  int ci=kchunk*32+kk, co=ntile*16+n;
  o[idx]=(h16)w[(co*Cc+ci)*9+tap];
}

__global__ __launch_bounds__(256) void k_pack1x1(const float* __restrict__ w, const float* __restrict__ scale,
                                                 h16* __restrict__ o, int Cin, int Cout){
  int idx=blockIdx.x*256+threadIdx.x;
  if(idx>=Cin*Cout) return;
  int kchunk, ntile, kk, n;
  frag_decode(idx, Cout>>4, kchunk, ntile, kk, n);
  int ci=kchunk*32+kk, co=ntile*16+n;
  float v=w[co*Cin+ci];
  if(scale) v*=scale[ci]*rsqrtf(1.f+1e-5f);
  o[idx]=(h16)v;
}

__global__ __launch_bounds__(256) void k_projconst(const float* __restrict__ w, const float* __restrict__ bb,
                                                   float* __restrict__ cst){
  int o=threadIdx.x;
  float s=0;
  for(int c=0;c<Cc;c++) s+=w[o*Cc+c]*bb[c];
  cst[o]=s;
}

// ---------------------------------------------------------------- implicit-GEMM 3x3 conv (WMMA)
// B fragments stream straight from global (frag-major, L2 resident); packed b32 LDS staging.
// mode 0: out = conv + vec1[co] + vec2[b,co]*in      mode 1: out = in*(vec1[b,co]+sigmoid(conv))
__global__ __launch_bounds__(256) void k_conv3(const float* __restrict__ in, const h16* __restrict__ Wp,
    const float* __restrict__ vec1, const float* __restrict__ vec2,
    float* __restrict__ out, int outCtot, int mode){
  const int tid=threadIdx.x, lane=tid&31, wid=tid>>5;
  const int gx=blockIdx.x; const int h=gx>>2, w0=(gx&3)*32;
  const int co0=blockIdx.y*64; const int b=blockIdx.z;
  __shared__ __align__(32) h16 Xp[34*32];     // input patch: cols w0-1..w0+32 x 32 cin ([col][k])
  const int msub=wid&1, nsub=wid>>1;
  const int ntile=(co0>>4)+nsub;
  // staging geometry (fixed per thread): main sweep cols 0..31, tail cols 32..33
  const int colA=tid&31, kpA=tid>>5;            // 32 cols x 8 kpairs
  const int ww2A=w0-1+colA;
  const int wclA=ww2A<0?0:(ww2A>127?127:ww2A);
  const int colB=32+(tid>>4), kpB=tid&15;       // tail: 2 cols x 16 kpairs (threads 0..31)
  const int ww2B=w0-1+colB;
  const int wclB=ww2B>127?127:ww2B;
  v8f acc = {};
  for(int dh=-1;dh<=1;dh++){
    int hh2=h+dh;
    bool hok=(hh2>=0)&&(hh2<Hc);
    int hcl=hok?hh2:0;
    float mA=(hok&&ww2A==wclA)?1.f:0.f;
    float mB=(hok&&ww2B==wclB)?1.f:0.f;
    for(int kc=0;kc<8;kc++){
      int cin0=kc*32;
      const float* src=in + (((size_t)b*Cc+cin0)*Hc+hcl)*Wd;
#pragma unroll
      for(int i=0;i<2;i++){                     // kpairs kpA and kpA+8
        int kk=(kpA+i*8)*2;
        float v0=src[(size_t)kk*NP+wclA]*mA;
        float v1=src[(size_t)(kk+1)*NP+wclA]*mA;
        h2v pk; pk[0]=(h16)v0; pk[1]=(h16)v1;
        *(h2v*)(&Xp[colA*32+kk])=pk;
      }
      if(tid<32){                               // cols 32,33
        int kk=kpB*2;
        float v0=src[(size_t)kk*NP+wclB]*mB;
        float v1=src[(size_t)(kk+1)*NP+wclB]*mB;
        h2v pk; pk[0]=(h16)v0; pk[1]=(h16)v1;
        *(h2v*)(&Xp[colB*32+kk])=pk;
      }
      if(kc<7) __builtin_prefetch(src+(size_t)32*NP+w0, 0, 1);   // next cin chunk
      __syncthreads();
#pragma unroll
      for(int t=0;t<3;t++){            // dw = t-1, col offset = m + t
        int tap=(dh+1)*3+t;
        v16h a=load_arow(Xp + (msub*16 + t)*32, lane, 32);
        v16h bb=load_frag16(Wp + ((size_t)((tap*8+kc)*16 + ntile)<<9), lane);
        acc=wmma16(a,bb,acc);
      }
      __syncthreads();
    }
  }
  int dn=lane&15, mb8=(lane>>4)*8;
  int co=co0+nsub*16+dn;
#pragma unroll
  for(int r=0;r<8;r++){
    int m=msub*16+mb8+r; int wpix=w0+m;
    size_t po=(((size_t)b*outCtot + co)*Hc+h)*Wd+wpix;
    size_t pi=(((size_t)b*Cc + co)*Hc+h)*Wd+wpix;
    float v;
    if(mode==0) v=acc[r]+vec1[co]+vec2[b*Cc+co]*in[pi];
    else        v=in[pi]*(vec1[b*Cc+co]+sigm(acc[r]));
    out[po]=v;
  }
}

// ---------------------------------------------------------------- 1x1 conv / pixel-GEMM (WMMA)
// 64x64 tile, 2 accumulators per wave sharing one A fragment.
// mode 0: plain; mode 1: + res[b,co,p]; mode 2: + cvec[co]
__global__ __launch_bounds__(256) void k_gemm1x1(const float* __restrict__ in, const h16* __restrict__ Wp,
    const float* __restrict__ res, const float* __restrict__ cvec, float* __restrict__ out,
    int Cin, int Cout, int mode){
  const int tid=threadIdx.x, lane=tid&31, wid=tid>>5;
  const int p0=blockIdx.x*64, co0=blockIdx.y*64, b=blockIdx.z;
  __shared__ __align__(32) h16 As[64*32];     // [m][k]
  const int msub=wid&3, nh=wid>>2;            // wave -> (msub, {nh, nh+2})
  const int ntile0=(co0>>4)+nh, ntile1=ntile0+2;
  const int nt=Cout>>4;
  const int mA=tid&63, kpA=tid>>6;            // staging: 64 m x 4 kpair groups
  v8f acc0 = {}, acc1 = {};
  for(int kc=0;kc<Cin/32;kc++){
    int cin0=kc*32;
    const float* src=in + ((size_t)b*Cin+cin0)*NP + p0;
#pragma unroll
    for(int i=0;i<4;i++){
      int kk=(kpA+i*4)*2;
      float v0=src[(size_t)kk*NP+mA];
      float v1=src[(size_t)(kk+1)*NP+mA];
      h2v pk; pk[0]=(h16)v0; pk[1]=(h16)v1;
      *(h2v*)(&As[mA*32+kk])=pk;
    }
    __syncthreads();
    v16h a=load_arow(&As[msub*16*32], lane, 32);
    v16h b0=load_frag16(Wp + ((size_t)(kc*nt + ntile0)<<9), lane);
    v16h b1=load_frag16(Wp + ((size_t)(kc*nt + ntile1)<<9), lane);
    acc0=wmma16(a,b0,acc0);
    acc1=wmma16(a,b1,acc1);
    __syncthreads();
  }
  int dn=lane&15, mb8=(lane>>4)*8;
#pragma unroll
  for(int s=0;s<2;s++){
    int co=co0+(nh+s*2)*16+dn;
#pragma unroll
    for(int r=0;r<8;r++){
      int p=p0+msub*16+mb8+r;
      float v=(s==0)?acc0[r]:acc1[r];
      if(mode==1) v+=res[((size_t)b*Cout+co)*NP+p];
      else if(mode==2) v+=cvec[co];
      out[((size_t)b*Cout+co)*NP+p]=v;
    }
  }
}

// ---------------------------------------------------------------- fused window attention
__global__ __launch_bounds__(256) void k_attn(const float* __restrict__ x, const h16* __restrict__ Wq,
    const float* __restrict__ rpb, float* __restrict__ o){
  const int tid=threadIdx.x, lane=tid&31, wid=tid>>5;
  const int win=blockIdx.x, b=blockIdx.y;
  const int hb=(win>>4)*8, wb=(win&15)*8;
  __shared__ __align__(32) h16 xw[64*256];    // window activations (A, row-major), 32 KB
  __shared__ __align__(32) h16 qL[64*32];     // q rows, K zero-padded to 32 (A operand)
  __shared__ __align__(32) h16 kswz[4*512];   // k^T tiles, B-frag-major, K zero-padded
  __shared__ __align__(32) h16 vswz[2*512];   // v K-chunks, B-frag-major
  __shared__ float sc[64*64];                 // scores f32; reused as f16 probs (A operand)
  const int dn=lane&15, mb8=(lane>>4)*8;

  for(int i=0;i<64;i++){
    int idx=i*256+tid; int cin=idx>>6, pix=idx&63;
    int hh2=hb+(pix>>3), ww2=wb+(pix&7);
    xw[pix*256+cin]=(h16)x[(((size_t)b*Cc+cin)*Hc+hh2)*Wd+ww2];
  }
  for(int i=tid;i<1024;i+=256) qL[(i>>4)*32 + 16 + (i&15)]=(h16)0.f;
  for(int i=tid;i<2048;i+=256){ if((i&15)>=8) kswz[i]=(h16)0.f; }
  __syncthreads();

  for(int h0=0;h0<16;h0++){
    // ---- qkv: 12 (mat,msub) tiles over 8 waves; barrier-free K loop
    v8f acc0 = {}, acc1 = {};
    const int pr0=wid, pr1=wid+8;
    const int mat0=pr0>>2, ms0=pr0&3;
    const int mat1=pr1>>2, ms1=pr1&3;
#pragma unroll
    for(int kc=0;kc<8;kc++){
      int cin0=kc*32;
      {
        v16h a=load_arow(&xw[ms0*16*256+cin0], lane, 256);
        v16h bb=load_frag16(Wq + ((size_t)(kc*48 + mat0*16 + h0)<<9), lane);
        acc0=wmma16(a,bb,acc0);
      }
      if(pr1<12){
        v16h a=load_arow(&xw[ms1*16*256+cin0], lane, 256);
        v16h bb=load_frag16(Wq + ((size_t)(kc*48 + mat1*16 + h0)<<9), lane);
        acc1=wmma16(a,bb,acc1);
      }
    }
    __syncthreads();   // qL/kswz/vswz free from previous head
    {
#pragma unroll
      for(int r=0;r<8;r++){
        int pix=ms0*16+mb8+r;
        if(mat0==0)      qL[pix*32+dn]=(h16)acc0[r];
        else if(mat0==1) kswz[(pix>>4)*512 + bswz(pix&15, dn)]=(h16)acc0[r];
        else             vswz[(pix>>5)*512 + bswz(dn, pix&31)]=(h16)acc0[r];
      }
    }
    if(pr1<12){
#pragma unroll
      for(int r=0;r<8;r++){
        int pix=ms1*16+mb8+r;
        if(mat1==0)      qL[pix*32+dn]=(h16)acc1[r];
        else if(mat1==1) kswz[(pix>>4)*512 + bswz(pix&15, dn)]=(h16)acc1[r];
        else             vswz[(pix>>5)*512 + bswz(dn, pix&31)]=(h16)acc1[r];
      }
    }
    __syncthreads();

    // ---- scores = q.k^T * hd^-0.5 + rel-pos bias; 16 tiles over 8 waves (2 each)
#pragma unroll
    for(int tt=0;tt<2;tt++){
      int t=wid+tt*8;
      int ti=t>>2, tj=t&3;
      v8f sa = {};
      v16h a=load_arow(&qL[(ti*16)*32], lane, 32);
      v16h bb=load_frag16(&kswz[tj*512], lane);
      sa=wmma16(a,bb,sa);
#pragma unroll
      for(int r=0;r<8;r++){
        int qi=ti*16+mb8+r, ki=tj*16+dn;
        int dr=(qi>>3)-(ki>>3)+7, dc=(qi&7)-(ki&7)+7;
        sc[qi*64+ki]=sa[r]*0.25f + rpb[(dr*15+dc)*16+h0];
      }
    }
    __syncthreads();

    // ---- softmax per row (threads 0..63: wave-uniform split), vectorized LDS I/O
    float rowreg[64];
    float inv=0.f;
    if(tid<64){
      const float4* rp=(const float4*)&sc[tid*64];
      float mx=-3.4e38f;
#pragma unroll
      for(int q4=0;q4<16;q4++){
        float4 t=rp[q4];
        rowreg[q4*4+0]=t.x; rowreg[q4*4+1]=t.y; rowreg[q4*4+2]=t.z; rowreg[q4*4+3]=t.w;
        mx=fmaxf(mx,fmaxf(fmaxf(t.x,t.y),fmaxf(t.z,t.w)));
      }
      float s=0;
#pragma unroll
      for(int j=0;j<64;j++){ rowreg[j]=__expf(rowreg[j]-mx); s+=rowreg[j]; }
      inv=1.f/s;
    }
    __syncthreads();
    h16* attnH=(h16*)sc;
    if(tid<64){
      h16* wr=&attnH[tid*64];
#pragma unroll
      for(int c8=0;c8<8;c8++){
        v8h pk;
#pragma unroll
        for(int j=0;j<8;j++) pk[j]=(h16)(rowreg[c8*8+j]*inv);
        *(v8h*)(wr+c8*8)=pk;
      }
    }
    __syncthreads();

    // ---- o = P.v : 4 tiles on waves 0-3, K=64 in two WMMA steps
    if(wid<4){
      v8f oc = {};
#pragma unroll
      for(int kc2=0;kc2<2;kc2++){
        v16h a=load_arow(&attnH[(wid*16)*64 + kc2*32], lane, 64);
        v16h bb=load_frag16(&vswz[kc2*512], lane);
        oc=wmma16(a,bb,oc);
      }
#pragma unroll
      for(int r=0;r<8;r++){
        int qi=wid*16+mb8+r;
        int hh2=hb+(qi>>3), ww2=wb+(qi&7);
        o[(((size_t)b*Cc + h0*16+dn)*Hc+hh2)*Wd+ww2]=oc[r];
      }
    }
    __syncthreads();
  }
}

// ---------------------------------------------------------------- out_sa = wx*ox + wy*oy
__global__ __launch_bounds__(256) void k_outsa(const float* __restrict__ o, const float* __restrict__ wx,
    const float* __restrict__ wy, float* __restrict__ cc){
  int idx=blockIdx.x*256+threadIdx.x;   // over B*C*NP
  int p=idx%NP; int bc=idx/NP; int b=bc/Cc, c=bc%Cc;
  int h=p>>7, w=p&127;
  const float* base=o+(size_t)bc*NP;
  float sx=0;
#pragma unroll
  for(int i=0;i<8;i++){ int r=h-3+i; int r2=(r==128)?126:r; if(r>=0&&r<=128) sx+=base[(r2<<7)+w]; }
  float sy=0;
#pragma unroll
  for(int i=0;i<8;i++){ int cl=w-3+i; int c2=(cl==128)?126:cl; if(cl>=0&&cl<=128) sy+=base[(h<<7)+c2]; }
  float v=wx[0]*sx*0.125f + wy[0]*sy*0.125f;
  cc[(((size_t)b*512)+256+c)*NP + p]=v;
}

// ---------------------------------------------------------------- depthwise 8x8 (reflect+zero pad)
__global__ __launch_bounds__(256) void k_dw(const float* __restrict__ y0, const float* __restrict__ wt,
                                            float* __restrict__ y1){
  int gb=blockIdx.x;             // (b*C + c)*64 + hpair
  int hp=gb&63; int bc=gb>>6;
  int c=bc%Cc;
  int h0=hp*2;
  __shared__ float wsm[64];
  __shared__ float tile[10*136];
  if(threadIdx.x<64) wsm[threadIdx.x]=wt[(size_t)c*64+threadIdx.x];
  const float* base=y0+(size_t)bc*NP;
  for(int i=threadIdx.x;i<10*136;i+=256){
    int rr=i/136, ccol=i%136;
    int r=h0-3+rr, cl=ccol-3;
    int r2=(r==128)?126:r, c2=(cl==128)?126:cl;
    int rc=r2<0?0:(r2>127?127:r2), ccl=c2<0?0:(c2>127?127:c2);
    float v=base[(rc<<7)+ccl];
    bool ok=(r>=0&&r<=128&&cl>=0&&cl<=128);
    tile[i]=ok?v:0.f;
  }
  __syncthreads();
  int h=h0+(threadIdx.x>>7), w=threadIdx.x&127;
  float s=0;
#pragma unroll
  for(int kh=0;kh<8;kh++)
#pragma unroll
    for(int kw=0;kw<8;kw++)
      s+=wsm[kh*8+kw]*tile[(h-h0+kh)*136 + w+kw];
  y1[(size_t)bc*NP + (h<<7)+w]=s;
}

// ---------------------------------------------------------------- launch
extern "C" void kernel_launch(void* const* d_in, const int* in_sizes, int n_in,
                              void* d_out, int out_size, void* d_ws, size_t ws_size,
                              hipStream_t stream) {
  (void)in_sizes; (void)n_in; (void)out_size; (void)ws_size;
  const float* x     = (const float*)d_in[0];
  const float* qkvw  = (const float*)d_in[1];
  const float* l1w   = (const float*)d_in[2];
  const float* l1g   = (const float*)d_in[3];
  const float* l1b   = (const float*)d_in[4];
  const float* l2w   = (const float*)d_in[5];
  const float* l2g   = (const float*)d_in[6];
  const float* l2b   = (const float*)d_in[7];
  const float* dww   = (const float*)d_in[8];
  const float* projg = (const float*)d_in[9];
  const float* projb = (const float*)d_in[10];
  const float* ppw   = (const float*)d_in[11];
  const float* wx    = (const float*)d_in[12];
  const float* wy    = (const float*)d_in[13];
  const float* freq  = (const float*)d_in[14];
  const float* fc1w  = (const float*)d_in[15];
  const float* fc1b  = (const float*)d_in[16];
  const float* fc2w  = (const float*)d_in[17];
  const float* fc2b  = (const float*)d_in[18];
  const float* rpb   = (const float*)d_in[19];
  const float* siw   = (const float*)d_in[20];
  const float* smw1  = (const float*)d_in[21];
  const float* smb1  = (const float*)d_in[22];
  const float* smw2  = (const float*)d_in[23];
  const float* smb2  = (const float*)d_in[24];
  const float* spw   = (const float*)d_in[25];
  const float* f1w   = (const float*)d_in[26];
  const float* f2w   = (const float*)d_in[27];
  float* out = (float*)d_out;

  char* base=(char*)d_ws; size_t off=0;
  auto alloc=[&](size_t bytes)->void*{ off=(off+255)&~(size_t)255; void* p=base+off; off+=bytes; return p; };
  const size_t U=(size_t)Bc*Cc*NP;   // elements per full tensor

  float* cc    =(float*)alloc(2*U*4);   // concat(out_cnn, out_sa); later x_res (first half)
  float* obuf  =(float*)alloc(U*4);     // attention o; later x_fused; later y_dw
  float* xib   =(float*)alloc(U*4);     // xi; later y0
  float* pool  =(float*)alloc(Bc*Cc*4);
  float* img   =(float*)alloc((size_t)Bc*NP*4);
  float* Rt    =(float*)alloc(128*128*4);
  float* St    =(float*)alloc(128*128*4);
  float* T1    =(float*)alloc((size_t)Bc*NP*4);
  float* T2    =(float*)alloc((size_t)Bc*NP*4);
  float* dv    =(float*)alloc((size_t)Bc*NP*4);
  float* ctx   =(float*)alloc(Bc*Cc*4);
  float* att   =(float*)alloc(Bc*Cc*4);
  float* alpha =(float*)alloc(Bc*Cc*4);
  float* biasc =(float*)alloc(Cc*4);
  float* zb    =(float*)alloc(Bc*Cc*4);
  float* scb   =(float*)alloc(Bc*Cc*4);
  float* cst   =(float*)alloc(Cc*4);
  h16* WcH =(h16*)alloc((size_t)9*Cc*Cc*2);
  h16* WsH =(h16*)alloc((size_t)9*Cc*Cc*2);
  h16* WqH =(h16*)alloc((size_t)Cc*768*2);
  h16* WiH =(h16*)alloc((size_t)512*Cc*2);
  h16* Wf1H=(h16*)alloc((size_t)Cc*Cc*2);
  h16* Wf2H=(h16*)alloc((size_t)Cc*Cc*2);
  h16* WpH =(h16*)alloc((size_t)Cc*Cc*2);

  dim3 t256(256);
  // stats + DCT context
  k_pool <<<dim3(Bc*Cc), t256, 0, stream>>>(x, pool);
  k_xm   <<<dim3(Bc*NP/256), t256, 0, stream>>>(x, img);
  k_trig <<<dim3(64), t256, 0, stream>>>(Rt, St);
  k_dct1 <<<dim3(Bc*128), dim3(128), 0, stream>>>(img, Rt, St, T1, T2);
  k_dct2 <<<dim3(Bc*128), dim3(128), 0, stream>>>(Rt, St, T1, T2, dv);
  k_embed<<<dim3(Bc), t256, 0, stream>>>(dv, pool, ctx);
  k_mlp_att<<<dim3(Bc), t256, 0, stream>>>(ctx, fc1w, fc1b, fc2w, fc2b, freq, att, alpha);
  // weight packing (frag-major)
  k_pack_cnn<<<dim3(9*Cc*Cc/256), t256, 0, stream>>>(l1w, l1g, l1b, l2w, l2g, l2b, freq, WcH, biasc);
  k_pack_s3 <<<dim3(9*Cc*Cc/256), t256, 0, stream>>>(spw, WsH);
  k_pack1x1 <<<dim3(Cc*768/256), t256, 0, stream>>>(qkvw, nullptr, WqH, Cc, 768);
  k_pack1x1 <<<dim3(512*Cc/256), t256, 0, stream>>>(siw, nullptr, WiH, 512, Cc);
  k_pack1x1 <<<dim3(Cc*Cc/256), t256, 0, stream>>>(f1w, nullptr, Wf1H, Cc, Cc);
  k_pack1x1 <<<dim3(Cc*Cc/256), t256, 0, stream>>>(f2w, nullptr, Wf2H, Cc, Cc);
  k_pack1x1 <<<dim3(Cc*Cc/256), t256, 0, stream>>>(ppw, projg, WpH, Cc, Cc);
  k_projconst<<<dim3(1), t256, 0, stream>>>(ppw, projb, cst);
  // CNN branch (two convs folded into one) -> cc[:, 0:256]
  k_conv3<<<dim3(Hc*4, Cc/64, Bc), t256, 0, stream>>>(x, WcH, biasc, alpha, cc, 512, 0);
  // fused window attention (qkv GEMM inside) -> obuf
  k_attn <<<dim3(NWIN, Bc), t256, 0, stream>>>(x, WqH, rpb, obuf);
  // directional pooling -> cc[:, 256:512]
  k_outsa<<<dim3((size_t)Bc*Cc*NP/256), t256, 0, stream>>>(obuf, wx, wy, cc);
  // SCARF
  k_gemm1x1<<<dim3(NP/64, Cc/64, Bc), t256, 0, stream>>>(cc, WiH, nullptr, nullptr, xib, 512, Cc, 0);   // xi
  k_zmax <<<dim3(Bc*Cc), t256, 0, stream>>>(xib, zb);
  k_mlp_scarf<<<dim3(Bc), t256, 0, stream>>>(zb, smw1, smb1, smw2, smb2, scb);
  k_conv3<<<dim3(Hc*4, Cc/64, Bc), t256, 0, stream>>>(xib, WsH, scb, nullptr, obuf, Cc, 1);             // x_fused
  k_gemm1x1<<<dim3(NP/64, Cc/64, Bc), t256, 0, stream>>>(obuf, Wf1H, xib, nullptr, cc, Cc, Cc, 1);      // x_res
  k_gemm1x1<<<dim3(NP/64, Cc/64, Bc), t256, 0, stream>>>(cc, Wf2H, nullptr, nullptr, xib, Cc, Cc, 0);   // y0
  // projection: depthwise 8x8 + BN-folded 1x1
  k_dw   <<<dim3(Bc*Cc*64), t256, 0, stream>>>(xib, dww, obuf);                                          // y_dw
  k_gemm1x1<<<dim3(NP/64, Cc/64, Bc), t256, 0, stream>>>(obuf, WpH, nullptr, cst, out, Cc, Cc, 2);       // final
}
